// CausalSelfAttention_15796889715283
// MI455X (gfx1250) — matrix operand order repaired and checked by reading, not compiled, not run on hardware
//
#include <hip/hip_runtime.h>

// ---------------------------------------------------------------------------
// CDNA5 (gfx1250) causal self-attention: bf16 WMMA GEMMs + flash attention.
// Fragment loads are 2 x ds_load_b128 per lane; global->LDS staging is
// batched within the iteration (no cross-WMMA register live ranges -> no
// spills), with global_prefetch_b8 of the next tile for latency hiding.
// ---------------------------------------------------------------------------

typedef __bf16 bf16;
typedef __attribute__((ext_vector_type(16))) __bf16 bf16x16;
typedef __attribute__((ext_vector_type(8)))  __bf16 bf16x8;
typedef __attribute__((ext_vector_type(8)))  float  f32x8;

#define TSEQ   4096
#define MDIM   768
#define NHEADS 6
#define HDIM   128

// ---------------------------------------------------------------------------
// WMMA fragment helpers (layouts per CDNA5 ISA §7.12.2, wave32)
// ---------------------------------------------------------------------------

// A-matrix 16x32 bf16, tile stored row-major [M][K] (ld multiple of 8):
// lane m = lane&15; lanes 0-15: f[0..7]=K0..7, f[8..15]=K16..23
//                   lanes 16-31: f[0..7]=K8..15, f[8..15]=K24..31
__device__ __forceinline__ bf16x16 load_frag_a(const bf16* lds, int ld,
                                               int row0, int k0, int lane) {
  const int m = row0 + (lane & 15);
  const int half = lane >> 4;
  const bf16* p = lds + m * ld + k0 + half * 8;
  bf16x8 lo = *(const bf16x8*)(p);
  bf16x8 hi = *(const bf16x8*)(p + 16);
  return __builtin_shufflevector(lo, hi, 0, 1, 2, 3, 4, 5, 6, 7,
                                 8, 9, 10, 11, 12, 13, 14, 15);
}

// B-matrix 32x16 bf16, tile stored TRANSPOSED row-major [N][K]:
// lane n = lane&15; lanes 0-15: f[0..7]=K0..7, f[8..15]=K16..23
//                   lanes 16-31: f[0..7]=K8..15, f[8..15]=K24..31 -> 2 x 16B loads.
__device__ __forceinline__ bf16x16 load_frag_bt(const bf16* lds, int ld,
                                                int n0, int k0, int lane) {
  const int n = n0 + (lane & 15);
  const int half = lane >> 4;
  const bf16* p = lds + n * ld + k0 + half * 8;
  bf16x8 lo = *(const bf16x8*)(p);
  bf16x8 hi = *(const bf16x8*)(p + 16);
  return __builtin_shufflevector(lo, hi, 0, 1, 2, 3, 4, 5, 6, 7,
                                 8, 9, 10, 11, 12, 13, 14, 15);
}

__device__ __forceinline__ f32x8 wmma_bf16(bf16x16 a, bf16x16 b, f32x8 c) {
  return __builtin_amdgcn_wmma_f32_16x16x32_bf16(
      /*neg_a=*/false, a, /*neg_b=*/false, b,
      /*c_mod=*/(short)0, c, /*reuse_a=*/false, /*reuse_b=*/false);
}

// C/D 16x16 f32: lane holds col = lane&15; VGPR r -> row r + 8*(lane>>4)
__device__ __forceinline__ void store_c(float* dst, int ld, int m0, int n0,
                                        f32x8 c, int lane) {
  const int n  = n0 + (lane & 15);
  const int mo = m0 + (lane >> 4) * 8;
#pragma unroll
  for (int r = 0; r < 8; ++r) dst[(size_t)(mo + r) * ld + n] = c[r];
}

// ---------------------------------------------------------------------------
// Elementwise / conversion kernels
// ---------------------------------------------------------------------------

__global__ void f32_to_bf16_kernel(const float* __restrict__ src,
                                   bf16* __restrict__ dst, int n) {
  int i = blockIdx.x * blockDim.x + threadIdx.x;
  if (i < n) dst[i] = (bf16)src[i];
}

// v = (1-lamb)*vraw + lamb*v_residual, to bf16, written TRANSPOSED per head:
// vt[(h*HDIM + d) * TSEQ + t]. Writes coalesced; strided reads are L2-resident
// (vraw+vres are 25 MB total vs 192 MB L2).
__global__ void vmix_t_kernel(const float* __restrict__ vraw,
                              const float* __restrict__ vres,
                              const float* __restrict__ lamb,
                              bf16* __restrict__ vt, int n) {
  int i = blockIdx.x * blockDim.x + threadIdx.x;
  if (i < n) {
    const int hd = i >> 12;          // 0 .. MDIM-1  (h*HDIM + d)
    const int t  = i & (TSEQ - 1);
    const size_t src = (size_t)t * MDIM + hd;
    float l = *lamb;
    vt[i] = (bf16)((1.0f - l) * vraw[src] + l * vres[src]);
  }
}

// Fused RMS-norm (over 128) + rotary embedding; one (t,head) row per block.
__global__ __launch_bounds__(128) void rmsnorm_rotary_kernel(
    const float* __restrict__ raw, const int* __restrict__ pos_id,
    bf16* __restrict__ out) {
  __shared__ float buf[HDIM];
  __shared__ float red[4];
  const int row = blockIdx.x;            // 0 .. TSEQ*NHEADS-1
  const int t = row / NHEADS, h = row % NHEADS;
  const int tid = threadIdx.x;
  const size_t base = (size_t)t * MDIM + h * HDIM;

  float v = raw[base + tid];
  float ss = v * v;
#pragma unroll
  for (int m = 16; m >= 1; m >>= 1) ss += __shfl_xor(ss, m, 32);
  if ((tid & 31) == 0) red[tid >> 5] = ss;
  __syncthreads();
  const float tot = red[0] + red[1] + red[2] + red[3];
  const float rn = rsqrtf(tot * (1.0f / HDIM) + 1.1920929e-7f);
  buf[tid] = v * rn;
  __syncthreads();

  if (tid < 64) {
    float x1 = buf[tid], x2 = buf[tid + 64];
    float c = 1.0f, s = 0.0f;
    if (tid < 32) {
      // angular_freq = (1/1024)^(tid/31); zeros for tid in [32,64)
      float af = __powf(1.0f / 1024.0f, (float)tid * (1.0f / 31.0f));
      float th = (float)pos_id[t] * af;
      __sincosf(th, &s, &c);
    }
    out[base + tid]      = (bf16)( x1 * c + x2 * s);
    out[base + tid + 64] = (bf16)(-x1 * s + x2 * c);
  }
}

// ---------------------------------------------------------------------------
// WMMA GEMM (fixed shape M=TSEQ, N=MDIM, K=MDIM):
//   C[M][N] f32 = A[M][K] bf16 @ Bt[N][K]^T bf16
// Bt row-major [N][K] is the ORIGINAL x@W.T weight orientation (no transpose).
// 4-wave workgroup, 128x64 tile (32x64 per wave), K-step 32.
// Staging is batched inside the iteration (short register live ranges, no
// spills); next k-tile is prefetched into cache one iteration ahead.
// ---------------------------------------------------------------------------
#define GT_BM 128
#define GT_BN 64
#define GT_BK 32
#define GT_LD (GT_BK + 8)
#define GT_NK (MDIM / GT_BK)   // 24 k-steps

__global__
__attribute__((amdgpu_flat_work_group_size(128, 128)))
__attribute__((amdgpu_waves_per_eu(1)))
void wmma_gemm_bf16_kernel(
    const bf16* __restrict__ A, const bf16* __restrict__ Bt,
    float* __restrict__ C) {
  __shared__ bf16 As[GT_BM][GT_LD];
  __shared__ bf16 Bs[GT_BN][GT_LD];
  const int tid = threadIdx.x, lane = tid & 31, wid = tid >> 5;
  const int m0 = blockIdx.x * GT_BM;
  const int n0 = blockIdx.y * GT_BN;

  // Per-thread staging chunk coordinates (constant across k-steps).
  // A tile: 128 rows x 4 chunks of 8 bf16 -> 512 chunks, 4/thread
  // B tile:  64 rows x 4 chunks           -> 256 chunks, 2/thread
  int arow[4], acol[4], brow[2], bcol[2];
#pragma unroll
  for (int j = 0; j < 4; ++j) {
    int c = tid + j * 128; arow[j] = c >> 2; acol[j] = (c & 3) * 8;
  }
#pragma unroll
  for (int j = 0; j < 2; ++j) {
    int c = tid + j * 128; brow[j] = c >> 2; bcol[j] = (c & 3) * 8;
  }

  const f32x8 zero = {0, 0, 0, 0, 0, 0, 0, 0};
  f32x8 acc[2][4];
#pragma unroll
  for (int i = 0; i < 2; ++i)
#pragma unroll
    for (int j = 0; j < 4; ++j) acc[i][j] = zero;

  // Warm the cache for the first tile.
  __builtin_prefetch(&A[(size_t)(m0 + arow[0]) * MDIM + acol[0]], 0, 1);
  __builtin_prefetch(&Bt[(size_t)(n0 + brow[0]) * MDIM + bcol[0]], 0, 1);

  for (int kt = 0; kt < GT_NK; ++kt) {
    const int k0 = kt * GT_BK;
    __syncthreads();   // previous tile fully consumed
    // Batched staging: all loads in flight, one wait, then LDS stores.
    {
      uint4 ar[4], br[2];
#pragma unroll
      for (int j = 0; j < 4; ++j)
        ar[j] = *(const uint4*)&A[(size_t)(m0 + arow[j]) * MDIM + k0 + acol[j]];
#pragma unroll
      for (int j = 0; j < 2; ++j)
        br[j] = *(const uint4*)&Bt[(size_t)(n0 + brow[j]) * MDIM + k0 + bcol[j]];
#pragma unroll
      for (int j = 0; j < 4; ++j) *(uint4*)&As[arow[j]][acol[j]] = ar[j];
#pragma unroll
      for (int j = 0; j < 2; ++j) *(uint4*)&Bs[brow[j]][bcol[j]] = br[j];
    }
    __syncthreads();
    if (kt + 1 < GT_NK) {  // global_prefetch_b8 of the next k-tile
      __builtin_prefetch(&A[(size_t)(m0 + arow[0]) * MDIM + k0 + GT_BK + acol[0]], 0, 1);
      __builtin_prefetch(&Bt[(size_t)(n0 + brow[0]) * MDIM + k0 + GT_BK + bcol[0]], 0, 1);
    }

    bf16x16 a0 = load_frag_a(&As[0][0], GT_LD, wid * 32, 0, lane);
    bf16x16 a1 = load_frag_a(&As[0][0], GT_LD, wid * 32 + 16, 0, lane);
#pragma unroll
    for (int nt = 0; nt < 4; ++nt) {
      bf16x16 b = load_frag_bt(&Bs[0][0], GT_LD, nt * 16, 0, lane);
      acc[0][nt] = wmma_bf16(a0, b, acc[0][nt]);
      acc[1][nt] = wmma_bf16(a1, b, acc[1][nt]);
    }
  }
#pragma unroll
  for (int i = 0; i < 2; ++i)
#pragma unroll
    for (int nt = 0; nt < 4; ++nt)
      store_c(C, MDIM, m0 + wid * 32 + i * 16, n0 + nt * 16, acc[i][nt], lane);
}

// ---------------------------------------------------------------------------
// Flash attention (causal), bf16 WMMA, f32 online softmax.
// Q,K in [T][MDIM] bf16; V in transposed [h][d][t] bf16; O f32 [T][MDIM].
// Grid: (NHEADS, TSEQ/64). 4 waves; each wave owns 16 query rows.
// K/V staging is batched: 8 global loads in flight, one wait, 8 LDS stores.
// ---------------------------------------------------------------------------
#define FA_BM 64
#define FA_BN 64
#define FA_LDQ (HDIM + 8)
#define FA_LDV (FA_BN + 8)

__global__
__attribute__((amdgpu_flat_work_group_size(128, 128)))
__attribute__((amdgpu_waves_per_eu(1)))
void flash_attn_bf16_kernel(
    const bf16* __restrict__ Q, const bf16* __restrict__ K,
    const bf16* __restrict__ Vt, float* __restrict__ O) {
  __shared__ bf16 Qs[FA_BM][FA_LDQ];          // [query][d]
  __shared__ bf16 Ks[FA_BN][FA_LDQ];          // [key][d]   == B^T for Q*K^T
  __shared__ bf16 Vs[HDIM][FA_LDV];           // [d][key]   == B^T for P*V
  __shared__ bf16 Ps[4][16][FA_LDV];          // per-wave P [m][key]

  const int h = blockIdx.x;
  const int mb = blockIdx.y;
  const int tid = threadIdx.x, lane = tid & 31, wid = tid >> 5;
  const int m0 = mb * FA_BM;
  const size_t colbase = (size_t)h * HDIM;
  const bf16* VtH = Vt + colbase * TSEQ;      // [d][t] rows of length TSEQ

  // Stage Q block once: 64 rows x 128 bf16 (batched 8 chunks/thread)
  {
    uint4 qr[8];
#pragma unroll
    for (int j = 0; j < 8; ++j) {
      int c = tid + j * 128, r = c >> 4, cc = (c & 15) * 8;
      qr[j] = *(const uint4*)&Q[(size_t)(m0 + r) * MDIM + colbase + cc];
    }
#pragma unroll
    for (int j = 0; j < 8; ++j) {
      int c = tid + j * 128, r = c >> 4, cc = (c & 15) * 8;
      *(uint4*)&Qs[r][cc] = qr[j];
    }
  }
  __syncthreads();
  bf16x16 qf[4];
#pragma unroll
  for (int kk = 0; kk < 4; ++kk)
    qf[kk] = load_frag_a(&Qs[0][0], FA_LDQ, wid * 16, kk * 32, lane);

  const f32x8 zero = {0, 0, 0, 0, 0, 0, 0, 0};
  f32x8 acc[8];
  float m_i[8], l_i[8];
#pragma unroll
  for (int i = 0; i < 8; ++i) { acc[i] = zero; m_i[i] = -1.0e30f; l_i[i] = 0.0f; }
  const float scale = 0.08838834764831845f;   // 1/sqrt(128)
  const int ncol = lane & 15;
  const int mrow_base = m0 + wid * 16 + (lane >> 4) * 8;

  for (int nb = 0; nb <= mb; ++nb) {
    const int n0 = nb * FA_BN;
    __syncthreads();
    // K block: 64 keys x 128 d — batch 8 loads, then 8 stores
    {
      uint4 kr[8];
#pragma unroll
      for (int j = 0; j < 8; ++j) {
        int c = tid + j * 128, r = c >> 4, cc = (c & 15) * 8;
        kr[j] = *(const uint4*)&K[(size_t)(n0 + r) * MDIM + colbase + cc];
      }
#pragma unroll
      for (int j = 0; j < 8; ++j) {
        int c = tid + j * 128, r = c >> 4, cc = (c & 15) * 8;
        *(uint4*)&Ks[r][cc] = kr[j];
      }
    }
    // V block (transposed in global): 128 d x 64 keys
    {
      uint4 vr[8];
#pragma unroll
      for (int j = 0; j < 8; ++j) {
        int c = tid + j * 128, r = c >> 3, cc = (c & 7) * 8;
        vr[j] = *(const uint4*)&VtH[(size_t)r * TSEQ + n0 + cc];
      }
#pragma unroll
      for (int j = 0; j < 8; ++j) {
        int c = tid + j * 128, r = c >> 3, cc = (c & 7) * 8;
        *(uint4*)&Vs[r][cc] = vr[j];
      }
    }
    __syncthreads();
    if (nb < mb) {  // global_prefetch_b8 of next K/V blocks
      __builtin_prefetch(&K[(size_t)(n0 + FA_BN) * MDIM + colbase], 0, 1);
      __builtin_prefetch(&VtH[n0 + FA_BN], 0, 1);
    }

    // S = Q * K^T
    f32x8 s[4];
#pragma unroll
    for (int nt = 0; nt < 4; ++nt) {
      s[nt] = zero;
#pragma unroll
      for (int kk = 0; kk < 4; ++kk) {
        bf16x16 bk = load_frag_bt(&Ks[0][0], FA_LDQ, nt * 16, kk * 32, lane);
        s[nt] = wmma_bf16(qf[kk], bk, s[nt]);
      }
    }
    // scale + causal mask
#pragma unroll
    for (int nt = 0; nt < 4; ++nt) {
      const int n = n0 + nt * 16 + ncol;
#pragma unroll
      for (int r = 0; r < 8; ++r) {
        float val = s[nt][r] * scale;
        if (n > mrow_base + r) val = -1.0e30f;
        s[nt][r] = val;
      }
    }
    // online softmax: rows live across 16 lanes of a half-wave
    float mnew[8], alpha[8];
#pragma unroll
    for (int r = 0; r < 8; ++r) {
      float mx = fmaxf(fmaxf(s[0][r], s[1][r]), fmaxf(s[2][r], s[3][r]));
#pragma unroll
      for (int d = 8; d >= 1; d >>= 1) mx = fmaxf(mx, __shfl_xor(mx, d, 32));
      mnew[r] = fmaxf(m_i[r], mx);
      alpha[r] = __expf(m_i[r] - mnew[r]);
      m_i[r] = mnew[r];
    }
#pragma unroll
    for (int nt = 0; nt < 4; ++nt)
#pragma unroll
      for (int r = 0; r < 8; ++r) s[nt][r] = __expf(s[nt][r] - mnew[r]);
#pragma unroll
    for (int r = 0; r < 8; ++r) {
      float sum = s[0][r] + s[1][r] + s[2][r] + s[3][r];
#pragma unroll
      for (int d = 8; d >= 1; d >>= 1) sum += __shfl_xor(sum, d, 32);
      l_i[r] = l_i[r] * alpha[r] + sum;
    }
#pragma unroll
    for (int o = 0; o < 8; ++o)
#pragma unroll
      for (int r = 0; r < 8; ++r) acc[o][r] *= alpha[r];

    // P (bf16) -> per-wave LDS region [m][key] to re-form A fragments
    {
      const int rr = (lane >> 4) * 8;
#pragma unroll
      for (int nt = 0; nt < 4; ++nt)
#pragma unroll
        for (int r = 0; r < 8; ++r)
          Ps[wid][rr + r][nt * 16 + ncol] = (bf16)s[nt][r];
    }
    // acc += P * V
#pragma unroll
    for (int kk = 0; kk < 2; ++kk) {
      bf16x16 pf = load_frag_a(&Ps[wid][0][0], FA_LDV, 0, kk * 32, lane);
#pragma unroll
      for (int dt = 0; dt < 8; ++dt) {
        bf16x16 bv = load_frag_bt(&Vs[0][0], FA_LDV, dt * 16, kk * 32, lane);
        acc[dt] = wmma_bf16(pf, bv, acc[dt]);
      }
    }
  }

  // epilogue: O /= l, store f32 [T][768]
  float rl[8];
#pragma unroll
  for (int r = 0; r < 8; ++r) rl[r] = 1.0f / l_i[r];
#pragma unroll
  for (int dt = 0; dt < 8; ++dt) {
    f32x8 o = acc[dt];
#pragma unroll
    for (int r = 0; r < 8; ++r) o[r] *= rl[r];
    store_c(O, MDIM, m0 + wid * 16, (int)colbase + dt * 16, o, lane);
  }
}

// ---------------------------------------------------------------------------
// Host orchestration
// ---------------------------------------------------------------------------
extern "C" void kernel_launch(void* const* d_in, const int* in_sizes, int n_in,
                              void* d_out, int out_size, void* d_ws, size_t ws_size,
                              hipStream_t stream) {
  const float* x    = (const float*)d_in[0];
  const float* vres = (const float*)d_in[1];
  const float* Wq   = (const float*)d_in[2];
  const float* Wk   = (const float*)d_in[3];
  const float* Wv   = (const float*)d_in[4];
  const float* Wo   = (const float*)d_in[5];
  const float* lamb = (const float*)d_in[6];
  const int*   pos  = (const int*)d_in[7];
  float* out = (float*)d_out;

  const size_t TD = (size_t)TSEQ * MDIM;   // 3,145,728
  const size_t DD = (size_t)MDIM * MDIM;

  char* wsp = (char*)d_ws;
  auto alloc = [&](size_t bytes) -> void* {
    void* p = (void*)wsp;
    wsp += (bytes + 255) & ~(size_t)255;
    return p;
  };
  bf16* xb    = (bf16*)alloc(TD * sizeof(bf16));
  bf16* Wqb   = (bf16*)alloc(DD * sizeof(bf16));   // original [o][d] orientation
  bf16* Wkb   = (bf16*)alloc(DD * sizeof(bf16));
  bf16* Wvb   = (bf16*)alloc(DD * sizeof(bf16));
  bf16* Wob   = (bf16*)alloc(DD * sizeof(bf16));
  float* qraw = (float*)alloc(TD * sizeof(float));
  float* kraw = (float*)alloc(TD * sizeof(float));
  float* vraw = (float*)alloc(TD * sizeof(float));
  bf16* qb    = (bf16*)alloc(TD * sizeof(bf16));
  bf16* kb    = (bf16*)alloc(TD * sizeof(bf16));
  bf16* vt    = (bf16*)alloc(TD * sizeof(bf16));   // [h][d][t]
  float* yatt = (float*)alloc(TD * sizeof(float));
  bf16* yb    = (bf16*)alloc(TD * sizeof(bf16));

  const int n = (int)TD;
  const int nw = (int)DD;
  // 1) convert x and weights to bf16 (weights stay in x@W.T orientation = B^T)
  f32_to_bf16_kernel<<<(n + 255) / 256, 256, 0, stream>>>(x, xb, n);
  f32_to_bf16_kernel<<<(nw + 255) / 256, 256, 0, stream>>>(Wq, Wqb, nw);
  f32_to_bf16_kernel<<<(nw + 255) / 256, 256, 0, stream>>>(Wk, Wkb, nw);
  f32_to_bf16_kernel<<<(nw + 255) / 256, 256, 0, stream>>>(Wv, Wvb, nw);
  f32_to_bf16_kernel<<<(nw + 255) / 256, 256, 0, stream>>>(Wo, Wob, nw);
  // 2) projections
  dim3 gg(TSEQ / GT_BM, MDIM / GT_BN);
  wmma_gemm_bf16_kernel<<<gg, 128, 0, stream>>>(xb, Wqb, qraw);
  wmma_gemm_bf16_kernel<<<gg, 128, 0, stream>>>(xb, Wkb, kraw);
  wmma_gemm_bf16_kernel<<<gg, 128, 0, stream>>>(xb, Wvb, vraw);
  // 3) rms-norm + rotary (q,k); value-residual mix -> transposed V
  rmsnorm_rotary_kernel<<<TSEQ * NHEADS, 128, 0, stream>>>(qraw, pos, qb);
  rmsnorm_rotary_kernel<<<TSEQ * NHEADS, 128, 0, stream>>>(kraw, pos, kb);
  vmix_t_kernel<<<(n + 255) / 256, 256, 0, stream>>>(vraw, vres, lamb, vt, n);
  // 4) causal flash attention
  dim3 ag(NHEADS, TSEQ / FA_BM);
  flash_attn_bf16_kernel<<<ag, 128, 0, stream>>>(qb, kb, vt, yatt);
  // 5) output projection straight into d_out
  f32_to_bf16_kernel<<<(n + 255) / 256, 256, 0, stream>>>(yatt, yb, n);
  wmma_gemm_bf16_kernel<<<gg, 128, 0, stream>>>(yb, Wob, out);
  // 6) second tuple element: v_residual passthrough
  hipMemcpyAsync(out + TD, vres, TD * sizeof(float),
                 hipMemcpyDeviceToDevice, stream);
}
